// GraphSage_63831803953183
// MI455X (gfx1250) — matrix operand order repaired and checked by reading, not compiled, not run on hardware
//
#include <hip/hip_runtime.h>

typedef __attribute__((ext_vector_type(16))) _Float16 v16h;
typedef __attribute__((ext_vector_type(8)))  _Float16 h8v;
typedef __attribute__((ext_vector_type(4)))  _Float16 h4v;
typedef __attribute__((ext_vector_type(8)))  float    v8f;

#define DIM      128
#define TWO_D    256
#define A_STRIDE 264   // 256 + 8 half pad to break LDS bank conflicts

// ---------------------------------------------------------------------------
// A fragment: 16x32 f16 (MxK), contiguous k order (same as B):
//   lane L: row m = L%16 ; half h = L/16
//   VGPR j  : K = 16*h + 2*j + p          (p = pair pos)
// ---------------------------------------------------------------------------
__device__ __forceinline__ v16h load_a_frag(const _Float16* At, int kt, int lane) {
    const int m  = lane & 15;
    const int h8 = (lane >> 4) << 3;                 // 0 or 8
    const _Float16* row = At + m * A_STRIDE + (kt << 5);
    v16h a;
#pragma unroll
    for (int j = 0; j < 8; ++j) {
        const int kb = (h8 << 1) + 2 * j;
        a[2 * j]     = row[kb];
        a[2 * j + 1] = row[kb + 1];
    }
    return a;
}

// ---------------------------------------------------------------------------
// B fragment: 32x16 f16 (KxN): lane L: col n = L%16, K = 16*(L/16)+2*j+p
// -> 16 contiguous halfs per lane from the cached 64 KB f16 weight copy.
// ---------------------------------------------------------------------------
__device__ __forceinline__ v16h load_b_frag(const _Float16* Wh, int ntile, int kt, int lane) {
    const int n  = (ntile << 4) + (lane & 15);
    const int kb = (kt << 5) + ((lane >> 4) << 4);   // kt*32 + {0,16}
    const _Float16* p = Wh + n * TWO_D + kb;
    v16h b;
#pragma unroll
    for (int e = 0; e < 16; ++e) b[e] = p[e];
    return b;
}

// ---- 4-wide row loads (one global_load_b128 / _b64 per lane) --------------
__device__ __forceinline__ float4 load4(const float* p) {
    return *(const float4*)p;
}
__device__ __forceinline__ float4 load4(const _Float16* p) {
    h4v v = *(const h4v*)p;
    return make_float4((float)v[0], (float)v[1], (float)v[2], (float)v[3]);
}

__global__ void convert_w_f16(const float* __restrict__ W, _Float16* __restrict__ Wh, int n) {
    int i = blockIdx.x * blockDim.x + threadIdx.x;
    if (i < n) Wh[i] = (_Float16)W[i];
}

// ---------------------------------------------------------------------------
// One SAGE layer. Block = 256 threads (8 wave32) handles 16 output rows.
// Gather: one WAVE per frontier row; each lane owns 4 feature dims, so every
// neighbor row is a single 512B (f32) / 256B (f16) wave-wide load. Neighbor
// indices are loaded once per lane and broadcast with readlane so gather
// addresses become SGPR bases. S is a template param -> full unroll -> all
// 26 row loads in flight (latency-bound HBM gather wants max MLP).
// ---------------------------------------------------------------------------
template <typename SrcT, int S>
__global__ void sage_layer_kernel(const SrcT* __restrict__ feat,
                                  const int* __restrict__ self_idx,
                                  const int* __restrict__ nbr,
                                  const _Float16* __restrict__ Wh,
                                  _Float16* __restrict__ h_out,
                                  float* __restrict__ out_f32,
                                  int nrows) {
    __shared__ _Float16 Atile[16 * A_STRIDE];   // 16 x 256 concat activations (f16)
    __shared__ float    zbuf[16][DIM];          // sigmoid outputs pre-normalize
    __shared__ float    psum[16][16];           // partial row sums
    __shared__ float    rscale[16];

    const int row0 = blockIdx.x << 4;
    if (row0 >= nrows) return;
    const int t    = threadIdx.x;
    const int lane = t & 31;

    // ---- Stage 1: gather + neighbor mean into LDS A-tile (f16) ----
    {
        const int d4   = lane << 2;             // 4 dims per lane
        const int wv   = t >> 5;                // wave id 0..7 = row-in-group
        const float invS = 1.0f / (float)S;
#pragma unroll
        for (int rp = 0; rp < 2; ++rp) {
            const int r = (rp << 3) + wv;
            const int m = row0 + r;
            const int si = self_idx ? self_idx[m] : m;
            const float4 hs = load4(feat + (size_t)si * DIM + d4);

            // one index per lane, broadcast below (S <= 32)
            const int* nb = nbr + (size_t)m * S;
            const int my_idx = (lane < S) ? nb[lane] : 0;

            float4 acc = make_float4(0.f, 0.f, 0.f, 0.f);
#pragma unroll
            for (int s = 0; s < S; ++s) {
                const int gi = __builtin_amdgcn_readlane(my_idx, s);  // SGPR base
                const float4 v = load4(feat + (size_t)gi * DIM + d4);
                acc.x += v.x; acc.y += v.y; acc.z += v.z; acc.w += v.w;
            }

            _Float16* arow = Atile + r * A_STRIDE;
            h4v hself, hmean;
            hself[0] = (_Float16)hs.x;  hself[1] = (_Float16)hs.y;
            hself[2] = (_Float16)hs.z;  hself[3] = (_Float16)hs.w;
            hmean[0] = (_Float16)(acc.x * invS); hmean[1] = (_Float16)(acc.y * invS);
            hmean[2] = (_Float16)(acc.z * invS); hmean[3] = (_Float16)(acc.w * invS);
            *(h4v*)(arow + d4)       = hself;   // ds_store_b64
            *(h4v*)(arow + DIM + d4) = hmean;   // ds_store_b64
        }
    }
    __syncthreads();

    // ---- Stage 2: [16x256] @ [256x16] per wave via v_wmma_f32_16x16x32_f16 ----
    const int ntile = t >> 5;                   // 8 waves -> 8 col tiles = 128 cols
    v8f c = {};
#pragma unroll
    for (int kt = 0; kt < 8; ++kt) {
        v16h a = load_a_frag(Atile, kt, lane);
        v16h b = load_b_frag(Wh, ntile, kt, lane);
        c = __builtin_amdgcn_wmma_f32_16x16x32_f16(false, a, false, b,
                                                   (short)0, c, false, false);
    }

    // ---- Stage 3: sigmoid, stage to LDS (C layout: VGPR v -> row v + 8*(L/16)) ----
    {
        const int col   = (ntile << 4) + (lane & 15);
        const int rbase = (lane >> 4) << 3;
#pragma unroll
        for (int v = 0; v < 8; ++v) {
            const float x = c[v];
            zbuf[rbase + v][col] = 1.0f / (1.0f + __expf(-x));
        }
    }
    __syncthreads();

    // ---- Stage 4: row-wise L2 norm, 256-thread two-stage reduction ----
    {
        const int rr = t >> 4, jj = t & 15;
        float p = 0.0f;
#pragma unroll
        for (int i = 0; i < 8; ++i) {
            const float z = zbuf[rr][(jj << 3) + i];
            p += z * z;
        }
        psum[rr][jj] = p;
    }
    __syncthreads();
    if (t < 16) {
        float s = 0.0f;
#pragma unroll
        for (int j = 0; j < 16; ++j) s += psum[t][j];
        rscale[t] = rsqrtf(s);
    }
    __syncthreads();

    // ---- Stage 5: scaled store, 8 contiguous cols per thread (b128 stores) ----
    {
        const int r  = t >> 4;
        const int c0 = (t & 15) << 3;
        const float sc = rscale[r];
        if (h_out) {
            h8v o;
#pragma unroll
            for (int i = 0; i < 8; ++i) o[i] = (_Float16)(zbuf[r][c0 + i] * sc);
            *(h8v*)(h_out + (size_t)(row0 + r) * DIM + c0) = o;       // 16B store
        } else {
            float4 o0, o1;
            o0.x = zbuf[r][c0 + 0] * sc; o0.y = zbuf[r][c0 + 1] * sc;
            o0.z = zbuf[r][c0 + 2] * sc; o0.w = zbuf[r][c0 + 3] * sc;
            o1.x = zbuf[r][c0 + 4] * sc; o1.y = zbuf[r][c0 + 5] * sc;
            o1.z = zbuf[r][c0 + 6] * sc; o1.w = zbuf[r][c0 + 7] * sc;
            float* op = out_f32 + (size_t)(row0 + r) * DIM + c0;
            *(float4*)(op)     = o0;                                  // 16B store
            *(float4*)(op + 4) = o1;                                  // 16B store
        }
    }
}

extern "C" void kernel_launch(void* const* d_in, const int* in_sizes, int n_in,
                              void* d_out, int out_size, void* d_ws, size_t ws_size,
                              hipStream_t stream) {
    (void)n_in; (void)out_size; (void)ws_size;

    const float* Z        = (const float*)d_in[0];
    const float* W1       = (const float*)d_in[1];
    const float* W2       = (const float*)d_in[2];
    const int*   node_idx = (const int*)d_in[3];
    const int*   nbr1     = (const int*)d_in[4];
    const int*   nbr2     = (const int*)d_in[5];

    constexpr int S1 = 25, S2 = 10;
    const int M = in_sizes[3];          // 45056
    const int B = in_sizes[5] / S2;     // 4096
    const int WELEMS = DIM * TWO_D;     // 32768 per weight matrix

    char* ws = (char*)d_ws;
    _Float16* Wh1 = (_Float16*)ws;
    _Float16* Wh2 = (_Float16*)(ws + (size_t)WELEMS * sizeof(_Float16));
    _Float16* h1  = (_Float16*)(ws + (size_t)2 * WELEMS * sizeof(_Float16));

    convert_w_f16<<<(WELEMS + 255) / 256, 256, 0, stream>>>(W1, Wh1, WELEMS);
    convert_w_f16<<<(WELEMS + 255) / 256, 256, 0, stream>>>(W2, Wh2, WELEMS);

    // Layer 1: 45056 frontier rows -> h1 (f16)
    sage_layer_kernel<float, S1><<<M / 16, 256, 0, stream>>>(
        Z, node_idx, nbr1, Wh1, h1, nullptr, M);

    // Layer 2: 4096 batch rows -> d_out (f32)
    sage_layer_kernel<_Float16, S2><<<B / 16, 256, 0, stream>>>(
        h1, nullptr, nbr2, Wh2, nullptr, (float*)d_out, B);
}
